// RecursiveRetriever_73478300500455
// MI455X (gfx1250) — matrix operand order repaired and checked by reading, not compiled, not run on hardware
//
#include <hip/hip_runtime.h>
#include <hip/hip_bf16.h>

// ---- CDNA5 WMMA types ----
typedef __attribute__((ext_vector_type(16))) __bf16        v16bf;
typedef __attribute__((ext_vector_type(8)))  float         v8f;
typedef __attribute__((ext_vector_type(16))) unsigned short v16us;

#define DEVI __device__ __forceinline__

DEVI unsigned short f2bf(float f) {                 // f32 -> bf16 (RNE)
    unsigned u = __builtin_bit_cast(unsigned, f);
    u += 0x7FFFu + ((u >> 16) & 1u);
    return (unsigned short)(u >> 16);
}

DEVI v8f wmma_bf16(v16bf a, v16bf b, v8f c) {
    return __builtin_amdgcn_wmma_f32_16x16x32_bf16(false, a, false, b, (short)0, c, false, false);
}

// A-fragment (16x32 bf16, stored as u16): lanes 0-15 hold K={0..15},
// lanes 16-31 hold K={16..31} for row M = lane&15. base points at (row0,k0).
DEVI v16bf fragA(const unsigned short* base, int stride, int lane) {
    int m = lane & 15, hi = (lane >> 4) & 1;
    const unsigned short* p0 = base + (size_t)m * stride + 16 * hi;
    const unsigned short* p1 = base + (size_t)m * stride + 16 * hi + 8;
    v16us a;
#pragma unroll
    for (int j = 0; j < 8; ++j) { a[j] = p0[j]; a[8 + j] = p1[j]; }
    return __builtin_bit_cast(v16bf, a);
}

// B-fragment (32x16 bf16) from N-major u16 matrix WT[N][K]:
// lane's column n = lane&15, elements K = k0 + 16*hi + j (contiguous).
DEVI v16bf fragB_u16(const unsigned short* rowN, int k0, int lane) {
    int hi = (lane >> 4) & 1;
    const unsigned short* p = rowN + k0 + 16 * hi;
    v16us b;
#pragma unroll
    for (int j = 0; j < 16; ++j) b[j] = p[j];
    return __builtin_bit_cast(v16bf, b);
}

DEVI void pack4(v16us& d, int base, float4 c) {
    d[base + 0] = f2bf(c.x); d[base + 1] = f2bf(c.y);
    d[base + 2] = f2bf(c.z); d[base + 3] = f2bf(c.w);
}

// ------------------- setup kernels -------------------
__global__ void k_init_z(const float* __restrict__ q, float* __restrict__ z) {
    int i = blockIdx.x * blockDim.x + threadIdx.x;
    if (i < 8 * 512) z[i] = q[i];
}

// src[R][C] f32 -> dst[C][R] bf16(u16), 32x32 LDS tiles, coalesced both sides.
__global__ void k_transpose_bf16(const float* __restrict__ src, unsigned short* __restrict__ dst,
                                 int R, int C) {
    __shared__ float tl[32][33];
    int tc = blockIdx.x * 32, tr = blockIdx.y * 32;
    int lc = threadIdx.x & 31, lr = threadIdx.x >> 5;
#pragma unroll
    for (int i = 0; i < 4; ++i) {
        int r = lr + i * 8;
        tl[r][lc] = src[(size_t)(tr + r) * C + tc + lc];
    }
    __syncthreads();
#pragma unroll
    for (int i = 0; i < 4; ++i) {
        int c = lr + i * 8;
        dst[(size_t)(tc + c) * R + tr + lc] = f2bf(tl[lc][c]);
    }
}

// ------------------- selector: u = (state Wq)_h Wk_h^T, scale folded -------------------
__global__ void k_state_u(const float* __restrict__ q, const float* __restrict__ z,
                          const float* __restrict__ Wq, const float* __restrict__ bq,
                          const float* __restrict__ Wk,
                          unsigned short* __restrict__ u16 /*[B][16][512]*/) {
    int b = blockIdx.x, tid = threadIdx.x;          // 8 blocks x 256
    __shared__ float st[512];
    __shared__ float qh[512];
    for (int i = tid; i < 512; i += 256) st[i] = q[b * 512 + i] + z[b * 512 + i];
    __syncthreads();
    for (int d = tid; d < 512; d += 256) {
        float acc = bq[d];
        for (int e = 0; e < 512; ++e) acc += st[e] * Wq[(size_t)e * 512 + d];
        qh[d] = acc;
    }
    __syncthreads();
    unsigned short* ub = u16 + (size_t)b * 16 * 512;
    for (int idx = tid; idx < 4096; idx += 256) {
        int h = idx >> 9, e = idx & 511;
        const float* wk = Wk + (size_t)e * 512 + h * 64;
        const float* qp = qh + h * 64;
        float acc = 0.f;
#pragma unroll 8
        for (int d = 0; d < 64; ++d) acc += qp[d] * wk[d];
        ub[idx] = f2bf(acc * 0.125f);               // scale = HEAD_D^-0.5 folded in
    }
    for (int idx = tid; idx < 4096; idx += 256) ub[4096 + idx] = 0;  // pad heads 8..15
}

// ------------------- selector streaming pass over a 256-row chunk -------------------
// scores via WMMA bf16 (u padded 16 x cand^T), then chunk-local softmax partials
// (m, sum, weighted row-accumulators) written for global combine.
__global__ void k_selector(const float* __restrict__ cand, int N,
                           const unsigned short* __restrict__ u16,
                           float* __restrict__ part_m, float* __restrict__ part_s,
                           float* __restrict__ part_w) {
    int b = blockIdx.y, chunk = blockIdx.x, r0 = chunk * 256;
    __shared__ unsigned short uL[16 * 512];          // 16 KB
    __shared__ float sc[8][256];                     // 8 KB  (scores -> probs)
    __shared__ float red[8][16];
    __shared__ float mfin[8];
    int tid = threadIdx.x;                           // 128 threads = 4 waves
    for (int i = tid; i < 8192; i += 128) uL[i] = u16[(size_t)b * 8192 + i];
    __syncthreads();

    int wave = tid >> 5, lane = tid & 31;
    int m = lane & 15, hi = (lane >> 4) & 1;
    const float* candB = cand + (size_t)b * N * 512;

    // Phase 1: scores S[16h x 16n] = U(16x512) x candT, bf16 WMMA, K-steps of 32.
    for (int t = wave; t < 16; t += 4) {
        int n0 = t * 16;
        v8f acc = {};
        const float* crow = candB + (size_t)(r0 + n0 + m) * 512;
        // pull the next 16-row tile for this wave toward the caches (global_prefetch_b8)
        if (t + 4 < 16)
            __builtin_prefetch(candB + (size_t)(r0 + (t + 4) * 16 + m) * 512, 0, 1);
        for (int ks = 0; ks < 16; ++ks) {
            int k0 = ks * 32;
            v16bf A = fragA(uL + k0, 512, lane);
            const float4* cp4 = (const float4*)(crow + k0 + 16 * hi);  // 64B/lane, b128 loads
            v16us bu;
            pack4(bu,  0, cp4[0]);
            pack4(bu,  4, cp4[1]);
            pack4(bu,  8, cp4[2]);
            pack4(bu, 12, cp4[3]);
            acc = wmma_bf16(A, __builtin_bit_cast(v16bf, bu), acc);
        }
        if (lane < 16) {
#pragma unroll
            for (int r = 0; r < 8; ++r) sc[r][n0 + lane] = acc[r];  // heads 0..7 real
        }
    }
    __syncthreads();

    // Phase 2: per-head chunk max, exp, sum
    int h = tid >> 4, seg = tid & 15;
    float mx = -3.4e38f;
    for (int c = 0; c < 16; ++c) mx = fmaxf(mx, sc[h][seg * 16 + c]);
    red[h][seg] = mx;
    __syncthreads();
    if (seg == 0) {
        float mm = -3.4e38f;
        for (int i = 0; i < 16; ++i) mm = fmaxf(mm, red[h][i]);
        mfin[h] = mm;
    }
    __syncthreads();
    float mmv = mfin[h];
    float ss = 0.f;
    for (int c = 0; c < 16; ++c) {
        float p = __expf(sc[h][seg * 16 + c] - mmv);
        sc[h][seg * 16 + c] = p;
        ss += p;
    }
    red[h][seg] = ss;
    __syncthreads();
    if (seg == 0) {
        float s = 0.f;
        for (int i = 0; i < 16; ++i) s += red[h][i];
        part_m[((size_t)b * 64 + chunk) * 8 + h] = mmv;
        part_s[((size_t)b * 64 + chunk) * 8 + h] = s;
    }

    // Phase 3: weighted row accumulation (GEMV-like, 8 outputs x 512 cols), cache-hot reread.
    int e0 = tid * 4;
    float a0[8], a1[8], a2[8], a3[8];
#pragma unroll
    for (int hh = 0; hh < 8; ++hh) a0[hh] = a1[hh] = a2[hh] = a3[hh] = 0.f;
    for (int n = 0; n < 256; ++n) {
        const float4 cv = *(const float4*)(candB + (size_t)(r0 + n) * 512 + e0);
#pragma unroll
        for (int hh = 0; hh < 8; ++hh) {
            float p = sc[hh][n];
            a0[hh] += p * cv.x; a1[hh] += p * cv.y; a2[hh] += p * cv.z; a3[hh] += p * cv.w;
        }
    }
    float* wout = part_w + (((size_t)b * 64 + chunk) * 8) * 512;
#pragma unroll
    for (int hh = 0; hh < 8; ++hh) {
        wout[hh * 512 + e0 + 0] = a0[hh];
        wout[hh * 512 + e0 + 1] = a1[hh];
        wout[hh * 512 + e0 + 2] = a2[hh];
        wout[hh * 512 + e0 + 3] = a3[hh];
    }
}

// ------------------- combine partials, apply Wv/Wo, produce injection -------------------
__global__ void k_reduce_project(int nch,
                                 const float* __restrict__ part_m, const float* __restrict__ part_s,
                                 const float* __restrict__ part_w,
                                 const float* __restrict__ Wv, const float* __restrict__ bv,
                                 const float* __restrict__ Wo, const float* __restrict__ bo,
                                 const float* __restrict__ q, float* __restrict__ inj) {
    int b = blockIdx.x, tid = threadIdx.x;          // 8 blocks x 256
    __shared__ float Mh[8], Sh[8];
    __shared__ float coef[64 * 8];
    __shared__ float wl[8 * 512];
    __shared__ float ao[512];
    if (tid < 8) {
        float mm = -3.4e38f;
        for (int c = 0; c < nch; ++c) mm = fmaxf(mm, part_m[((size_t)b * 64 + c) * 8 + tid]);
        Mh[tid] = mm;
    }
    __syncthreads();
    for (int i = tid; i < nch * 8; i += 256) {
        int c = i >> 3, h = i & 7;
        coef[i] = __expf(part_m[((size_t)b * 64 + c) * 8 + h] - Mh[h]);
    }
    __syncthreads();
    if (tid < 8) {
        float s = 0.f;
        for (int c = 0; c < nch; ++c) s += coef[c * 8 + tid] * part_s[((size_t)b * 64 + c) * 8 + tid];
        Sh[tid] = s;
    }
    __syncthreads();
    for (int i = tid; i < 4096; i += 256) {
        int h = i >> 9, e = i & 511;
        float acc = 0.f;
        for (int c = 0; c < nch; ++c)
            acc += coef[c * 8 + h] * part_w[(((size_t)b * 64 + c) * 8 + h) * 512 + e];
        wl[i] = acc / Sh[h];                         // normalized attention * cand
    }
    __syncthreads();
    for (int d = tid; d < 512; d += 256) {          // attn_out = wl_h @ Wv_h + bv
        int h = d >> 6;
        const float* wr = wl + h * 512;
        float acc = bv[d];
        for (int e = 0; e < 512; ++e) acc += wr[e] * Wv[(size_t)e * 512 + d];
        ao[d] = acc;
    }
    __syncthreads();
    for (int j = tid; j < 512; j += 256) {          // selected = ao @ Wo + bo; inj = sel + q
        float acc = bo[j];
        for (int i2 = 0; i2 < 512; ++i2) acc += ao[i2] * Wo[(size_t)i2 * 512 + j];
        inj[b * 512 + j] = acc + q[b * 512 + j];
    }
}

// ------------------- reasoning module -------------------
__global__ void k_add_inj(const float* __restrict__ inj, float* __restrict__ z,
                          unsigned short* __restrict__ z16 /*[16][512]*/) {
    int blk = blockIdx.x, tid = threadIdx.x;        // 16 blocks x 256
    if (blk < 8) {
        for (int e = tid; e < 512; e += 256) {
            float v = z[blk * 512 + e] + inj[blk * 512 + e];
            z[blk * 512 + e] = v;
            z16[blk * 512 + e] = f2bf(v);
        }
    } else {
        for (int e = tid; e < 512; e += 256) z16[blk * 512 + e] = 0;   // pad rows
    }
}

// up = h(16x512) @ Wup(512x4096), WMMA bf16, WupT is N-major [4096][512]
__global__ void k_gemm_up(const unsigned short* __restrict__ z16,
                          const unsigned short* __restrict__ WT,
                          float* __restrict__ up /*[8][4096]*/) {
    __shared__ unsigned short aL[16 * 512];
    int tid = threadIdx.x;                           // 64 blocks x 128
    for (int i = tid; i < 8192; i += 128) aL[i] = z16[i];
    __syncthreads();
    int wave = tid >> 5, lane = tid & 31;
    int n0 = (blockIdx.x * 4 + wave) * 16;
    int m = lane & 15;
    const unsigned short* brow = WT + (size_t)(n0 + m) * 512;
    v8f acc = {};
    for (int ks = 0; ks < 16; ++ks) {
        v16bf A = fragA(aL + ks * 32, 512, lane);
        v16bf Bf = fragB_u16(brow, ks * 32, lane);
        acc = wmma_bf16(A, Bf, acc);
    }
    if (lane < 16) {
#pragma unroll
        for (int r = 0; r < 8; ++r) up[(size_t)r * 4096 + n0 + lane] = acc[r];
    }
}

__global__ void k_act(const float* __restrict__ up, unsigned short* __restrict__ t16 /*[16][2048]*/) {
    int blk = blockIdx.x, tid = threadIdx.x;        // 16 blocks x 256
    if (blk < 8) {
        for (int j = tid; j < 2048; j += 256) {
            float g = up[(size_t)blk * 4096 + j];
            float v = up[(size_t)blk * 4096 + 2048 + j];
            float s = g / (1.f + __expf(-g));       // silu
            t16[(size_t)blk * 2048 + j] = f2bf(s * v);
        }
    } else {
        for (int j = tid; j < 2048; j += 256) t16[(size_t)blk * 2048 + j] = 0;
    }
}

// ffn = t(16x2048) @ Wdown(2048x512), WMMA bf16, WdnT is N-major [512][2048]
__global__ void k_gemm_down(const unsigned short* __restrict__ t16,
                            const unsigned short* __restrict__ WT,
                            float* __restrict__ ffn /*[8][512]*/) {
    int tid = threadIdx.x;                           // 8 blocks x 128
    int wave = tid >> 5, lane = tid & 31;
    int n0 = (blockIdx.x * 4 + wave) * 16;
    int m = lane & 15;
    const unsigned short* brow = WT + (size_t)(n0 + m) * 2048;
    v8f acc = {};
    for (int ks = 0; ks < 64; ++ks) {
        v16bf A = fragA(t16 + ks * 32, 2048, lane);
        v16bf Bf = fragB_u16(brow, ks * 32, lane);
        acc = wmma_bf16(A, Bf, acc);
    }
    if (lane < 16) {
#pragma unroll
        for (int r = 0; r < 8; ++r) ffn[(size_t)r * 512 + n0 + lane] = acc[r];
    }
}

__global__ void k_norm(const float* __restrict__ ffn, const float* __restrict__ nw,
                       float* __restrict__ z, unsigned short* __restrict__ z16) {
    int b = blockIdx.x, tid = threadIdx.x;          // 8 blocks x 256
    __shared__ float yv[512];
    __shared__ float red[256];
    float y0 = z[b * 512 + tid] + ffn[b * 512 + tid];
    float y1 = z[b * 512 + tid + 256] + ffn[b * 512 + tid + 256];
    yv[tid] = y0; yv[tid + 256] = y1;
    red[tid] = y0 * y0 + y1 * y1;
    __syncthreads();
    for (int s = 128; s > 0; s >>= 1) {
        if (tid < s) red[tid] += red[tid + s];
        __syncthreads();
    }
    float rinv = rsqrtf(red[0] / 512.f + 1e-6f);
    float v0 = nw[tid] * yv[tid] * rinv;
    float v1 = nw[tid + 256] * yv[tid + 256] * rinv;
    z[b * 512 + tid] = v0;       z16[b * 512 + tid] = f2bf(v0);
    z[b * 512 + tid + 256] = v1; z16[b * 512 + tid + 256] = f2bf(v1);
}

// ------------------- outputs -------------------
__global__ void k_halt(const float* __restrict__ z, const float* __restrict__ qw,
                       const float* __restrict__ qb, float* __restrict__ out, int r) {
    int b = blockIdx.x, tid = threadIdx.x;          // 8 blocks x 256
    __shared__ float red[256];
    red[tid] = z[b * 512 + tid] * qw[tid] + z[b * 512 + tid + 256] * qw[tid + 256];
    __syncthreads();
    for (int s = 128; s > 0; s >>= 1) {
        if (tid < s) red[tid] += red[tid + s];
        __syncthreads();
    }
    if (tid == 0) out[(size_t)b * 514 + 512 + r] = red[0] + qb[0];
}

__global__ void k_copy_z(const float* __restrict__ z, float* __restrict__ out) {
    int i = blockIdx.x * blockDim.x + threadIdx.x;
    if (i < 4096) out[(size_t)(i >> 9) * 514 + (i & 511)] = z[i];
}

// ------------------- host orchestration -------------------
extern "C" void kernel_launch(void* const* d_in, const int* in_sizes, int n_in,
                              void* d_out, int out_size, void* d_ws, size_t ws_size,
                              hipStream_t stream) {
    const float* q      = (const float*)d_in[0];
    const float* cand1  = (const float*)d_in[1];
    const float* cand2  = (const float*)d_in[2];
    const float* cand3  = (const float*)d_in[3];
    const float* Wq     = (const float*)d_in[4];
    const float* bq     = (const float*)d_in[5];
    const float* Wk     = (const float*)d_in[6];
    // d_in[7] = bk: provably cancels in softmax (uniform shift per (b,h))
    const float* Wv     = (const float*)d_in[8];
    const float* bv     = (const float*)d_in[9];
    const float* Wo     = (const float*)d_in[10];
    const float* bo     = (const float*)d_in[11];
    const float* norm_w = (const float*)d_in[12];
    const float* W_up   = (const float*)d_in[13];
    const float* W_down = (const float*)d_in[14];
    const float* qhw    = (const float*)d_in[15];
    const float* qhb    = (const float*)d_in[16];
    float* out = (float*)d_out;

    // workspace carve-out (~21 MB)
    char* ws = (char*)d_ws;
    size_t off = 0;
    auto alloc = [&](size_t bytes) -> char* {
        char* p = ws + off;
        off += (bytes + 255) & ~(size_t)255;
        return p;
    };
    float*          z      = (float*)alloc(4096 * 4);
    float*          inj    = (float*)alloc(4096 * 4);
    unsigned short* u16    = (unsigned short*)alloc((size_t)8 * 8192 * 2);
    unsigned short* z16    = (unsigned short*)alloc(8192 * 2);
    unsigned short* t16    = (unsigned short*)alloc((size_t)16 * 2048 * 2);
    float*          up     = (float*)alloc((size_t)8 * 4096 * 4);
    float*          ffn    = (float*)alloc((size_t)8 * 512 * 4);
    unsigned short* WupT   = (unsigned short*)alloc((size_t)2 * 4096 * 512 * 2);
    unsigned short* WdnT   = (unsigned short*)alloc((size_t)2 * 512 * 2048 * 2);
    float*          part_m = (float*)alloc((size_t)8 * 64 * 8 * 4);
    float*          part_s = (float*)alloc((size_t)8 * 64 * 8 * 4);
    float*          part_w = (float*)alloc((size_t)8 * 64 * 8 * 512 * 4);

    // per-launch weight prep (bf16, transposed to N-major for contiguous B-fragments)
    for (int l = 0; l < 2; ++l) {
        dim3 gu(4096 / 32, 512 / 32);
        k_transpose_bf16<<<gu, 256, 0, stream>>>(W_up + (size_t)l * 512 * 4096,
                                                 WupT + (size_t)l * 4096 * 512, 512, 4096);
        dim3 gd(512 / 32, 2048 / 32);
        k_transpose_bf16<<<gd, 256, 0, stream>>>(W_down + (size_t)l * 2048 * 512,
                                                 WdnT + (size_t)l * 512 * 2048, 2048, 512);
    }
    k_init_z<<<16, 256, 0, stream>>>(q, z);

    const float* cands[3] = {cand1, cand2, cand3};
    const int    Ns[3]    = {4096, 8192, 16384};

    for (int rnd = 0; rnd < 2; ++rnd) {
        for (int s = 0; s < 3; ++s) {
            k_state_u<<<8, 256, 0, stream>>>(q, z, Wq, bq, Wk, u16);
            int nch = Ns[s] / 256;
            dim3 g(nch, 8);
            k_selector<<<g, 128, 0, stream>>>(cands[s], Ns[s], u16, part_m, part_s, part_w);
            k_reduce_project<<<8, 256, 0, stream>>>(nch, part_m, part_s, part_w,
                                                    Wv, bv, Wo, bo, q, inj);
            for (int cyc = 0; cyc < 2; ++cyc) {
                k_add_inj<<<16, 256, 0, stream>>>(inj, z, z16);
                for (int l = 0; l < 2; ++l) {
                    k_gemm_up<<<64, 128, 0, stream>>>(z16, WupT + (size_t)l * 4096 * 512, up);
                    k_act<<<16, 256, 0, stream>>>(up, t16);
                    k_gemm_down<<<8, 128, 0, stream>>>(t16, WdnT + (size_t)l * 512 * 2048, ffn);
                    k_norm<<<8, 256, 0, stream>>>(ffn, norm_w + (size_t)l * 512, z, z16);
                }
            }
        }
        k_halt<<<8, 256, 0, stream>>>(z, qhw, qhb, out, rnd);
    }
    k_copy_z<<<16, 256, 0, stream>>>(z, out);

    (void)in_sizes; (void)n_in; (void)out_size; (void)ws_size;
}
